// IndRNN_83571473645654
// MI455X (gfx1250) — compile-verified, matrix operand order checked by reading, not run on hardware
//
#include <hip/hip_runtime.h>

// IndRNN: SEQ=1024, BATCH=64, IN=512, HID=1024, NLAYER=4
// inputs: x(S,B,IN) f32, W_ih0(H,IN) f32, W_ih(3,H,H) f32, w_hh(4,H) f32, b_ih(4,H) f32
// output: out(S,B,H) f32 ++ h_last(B,H) f32

#define SEQ    1024
#define BATCH  64
#define DIN    512
#define HID    1024

typedef __attribute__((ext_vector_type(16))) _Float16 v16h;
typedef __attribute__((ext_vector_type(8)))  _Float16 v8h;
typedef __attribute__((ext_vector_type(8)))  float    v8f;

// ------------------------------------------------ async-to-LDS feature gate -
#if defined(__AMDGCN__) && \
    __has_builtin(__builtin_amdgcn_global_load_async_to_lds_b128)
#define USE_ASYNC_LDS 1
#else
#define USE_ASYNC_LDS 0
#endif

#if USE_ASYNC_LDS
// builtin signature (from hipcc diagnostic): param0 = <4 x i32> addrspace(1)*,
// param1 = <4 x i32> addrspace(3)*, then imm offset, imm cpol.
typedef int v4i_vs __attribute__((__vector_size__(16)));
typedef __attribute__((address_space(1))) v4i_vs* gv4_ptr;
typedef __attribute__((address_space(3))) v4i_vs* lv4_ptr;

__device__ __forceinline__ gv4_ptr as_g(const void* p) {
  return (gv4_ptr)(uintptr_t)p;            // generic == global for global mem
}
__device__ __forceinline__ lv4_ptr as_l(void* p) {
  return (lv4_ptr)(uint32_t)(uintptr_t)p;  // LDS offset = low 32 bits (aperture rule)
}
__device__ __forceinline__ void wait_async0() {
#if __has_builtin(__builtin_amdgcn_s_wait_asynccnt)
  __builtin_amdgcn_s_wait_asynccnt(0);
#else
  asm volatile("s_wait_asynccnt 0x0" ::: "memory");
#endif
}
#endif

// ---------------------------------------------------------------- convert ---
__global__ __launch_bounds__(256) void cvt_f32_to_f16(
    const float* __restrict__ src, _Float16* __restrict__ dst, int n) {
  int i = blockIdx.x * blockDim.x + threadIdx.x;
  int stride = gridDim.x * blockDim.x;
  for (; i < n; i += stride) dst[i] = (_Float16)src[i];
}

// ------------------------------------------------------------------- GEMM ---
// C[M,N] (f32) = A[M,K](f16) * W[N,K](f16)^T + bias[N]
// BM=BN=128, BK=32; 256 threads = 8 waves; wave grid 4(m) x 2(n),
// each wave computes 32x64 = 2x4 fragments of v_wmma_f32_16x16x32_f16.
// Double-buffered LDS; tile k+1 DMA'd (async-to-LDS) while tile k computes.
#define BM  128
#define BN  128
#define BK  32
#define LDT 40   // padded LDS row stride in halves (80B, 16B-aligned chunks)

__global__ __launch_bounds__(256) void wmma_gemm_f16(
    const _Float16* __restrict__ A,   // M x K, row-major
    const _Float16* __restrict__ W,   // N x K, row-major (so B = W^T)
    const float*    __restrict__ bias,// N
    float*          __restrict__ C,   // M x N
    int M, int N, int K)
{
  __shared__ _Float16 lA[2][BM * LDT];
  __shared__ _Float16 lB[2][BN * LDT];

  const int tid  = threadIdx.x;
  const int lane = tid & 31;
  const int wave = tid >> 5;
  const int wm   = wave & 3;        // 0..3 -> 32 rows each
  const int wn   = wave >> 2;       // 0..1 -> 64 cols each

  const int bm = blockIdx.y * BM;
  const int bn = blockIdx.x * BN;

  const int l16   = lane & 15;
  const int hgrp  = lane >> 4;      // 0 or 1
  const int khalf = hgrp * 8;       // K sub-chunk base per ISA 16-bit layout

  v8f acc[2][4];
  #pragma unroll
  for (int mf = 0; mf < 2; ++mf)
    #pragma unroll
    for (int nf = 0; nf < 4; ++nf) {
      float bv = bias[bn + wn * 64 + nf * 16 + l16];
      acc[mf][nf] = (v8f){bv, bv, bv, bv, bv, bv, bv, bv};
    }

  // 128 rows x 32 halves = 512 chunks of 8 halves (16B); 2 chunks/thread/tile
#if USE_ASYNC_LDS
  auto issue_tile = [&](int kt, int buf) {
    #pragma unroll
    for (int cc = 0; cc < 2; ++cc) {
      int ch  = tid + cc * 256;
      int row = ch >> 2;
      int ko  = (ch & 3) * 8;
      __builtin_amdgcn_global_load_async_to_lds_b128(
          as_g(&A[(size_t)(bm + row) * K + kt + ko]),
          as_l(&lA[buf][row * LDT + ko]), 0, 0);
      __builtin_amdgcn_global_load_async_to_lds_b128(
          as_g(&W[(size_t)(bn + row) * K + kt + ko]),
          as_l(&lB[buf][row * LDT + ko]), 0, 0);
    }
  };
  issue_tile(0, 0);
  int buf = 0;
#endif

  for (int kt = 0; kt < K; kt += BK) {
#if USE_ASYNC_LDS
    wait_async0();        // my tile-`buf` DMAs complete
    __syncthreads();      // everyone's complete; also: all reads of buf^1 done
    if (kt + BK < K) issue_tile(kt + BK, buf ^ 1);   // overlap next tile DMA
#else
    const int buf = 0;
    __syncthreads();
    #pragma unroll
    for (int cc = 0; cc < 2; ++cc) {
      int ch  = tid + cc * 256;
      int row = ch >> 2;
      int ko  = (ch & 3) * 8;
      *(v8h*)&lA[0][row * LDT + ko] =
          *(const v8h*)&A[(size_t)(bm + row) * K + kt + ko];
      *(v8h*)&lB[0][row * LDT + ko] =
          *(const v8h*)&W[(size_t)(bn + row) * K + kt + ko];
    }
    __syncthreads();
#endif

    // A fragment (16x32 f16): lane<16 -> row m, K 0-7 & 16-23; lane>=16 -> K 8-15 & 24-31
    v16h afrag[2], bfrag[4];
    #pragma unroll
    for (int mf = 0; mf < 2; ++mf) {
      const _Float16* p = &lA[buf][(wm * 32 + mf * 16 + l16) * LDT + khalf];
      v8h lo = *(const v8h*)p;
      v8h hi = *(const v8h*)(p + 16);
      afrag[mf] = __builtin_shufflevector(lo, hi,
          0,1,2,3,4,5,6,7,8,9,10,11,12,13,14,15);
    }
    #pragma unroll
    for (int nf = 0; nf < 4; ++nf) {
      const _Float16* p = &lB[buf][(wn * 64 + nf * 16 + l16) * LDT + khalf];
      v8h lo = *(const v8h*)p;
      v8h hi = *(const v8h*)(p + 16);
      bfrag[nf] = __builtin_shufflevector(lo, hi,
          0,1,2,3,4,5,6,7,8,9,10,11,12,13,14,15);
    }

    #pragma unroll
    for (int mf = 0; mf < 2; ++mf)
      #pragma unroll
      for (int nf = 0; nf < 4; ++nf)
        acc[mf][nf] = __builtin_amdgcn_wmma_f32_16x16x32_f16(
            false, afrag[mf], false, bfrag[nf],
            (short)0, acc[mf][nf], false, false);

#if USE_ASYNC_LDS
    buf ^= 1;
#endif
  }

  // C/D layout: VGPR i, lanes 0-15 -> M=i, N=lane; lanes 16-31 -> M=8+i, N=lane-16
  const int rbase = bm + wm * 32 + 8 * hgrp;
  const int cbase = bn + wn * 64 + l16;
  #pragma unroll
  for (int mf = 0; mf < 2; ++mf)
    #pragma unroll
    for (int nf = 0; nf < 4; ++nf)
      #pragma unroll
      for (int i = 0; i < 8; ++i)
        C[(size_t)(rbase + mf * 16 + i) * N + cbase + nf * 16] = acc[mf][nf][i];
}

// ------------------------------------------------------------------- scan ---
// One thread per (b,h). pre laid out (s, b, h) with h fastest -> coalesced.
__global__ __launch_bounds__(256) void indrnn_scan(
    const float* __restrict__ pre,      // S x (B*H)
    const float* __restrict__ whh,      // H
    _Float16*    __restrict__ act_next, // S x (B*H) f16, or null
    float*       __restrict__ out_f32,  // S x (B*H) f32, or null
    float*       __restrict__ hlast)    // B*H f32, or null
{
  const int tid = blockIdx.x * blockDim.x + threadIdx.x;  // b*HID + h
  const float w = whh[tid & (HID - 1)];
  const int stride = BATCH * HID;
  float h = 0.f;
  #pragma unroll 4
  for (int s = 0; s < SEQ; ++s) {
    float p = pre[(size_t)s * stride + tid];
    h = fmaxf(fmaf(w, h, p), 0.f);
    if (act_next) act_next[(size_t)s * stride + tid] = (_Float16)h;
    if (out_f32)  out_f32[(size_t)s * stride + tid] = h;
  }
  if (hlast) hlast[tid] = h;
}

// ----------------------------------------------------------------- launch ---
extern "C" void kernel_launch(void* const* d_in, const int* in_sizes, int n_in,
                              void* d_out, int out_size, void* d_ws, size_t ws_size,
                              hipStream_t stream) {
  const float* x    = (const float*)d_in[0];
  const float* Wih0 = (const float*)d_in[1];
  const float* Wih  = (const float*)d_in[2];
  const float* whh  = (const float*)d_in[3];
  const float* bih  = (const float*)d_in[4];
  float* out = (float*)d_out;
  (void)in_sizes; (void)n_in; (void)out_size; (void)ws_size;

  char* ws = (char*)d_ws;
  size_t off = 0;
  _Float16* xh   = (_Float16*)(ws + off); off += (size_t)SEQ * BATCH * DIN * 2;
  _Float16* actA = (_Float16*)(ws + off); off += (size_t)SEQ * BATCH * HID * 2;
  _Float16* actB = (_Float16*)(ws + off); off += (size_t)SEQ * BATCH * HID * 2;
  float*    pre  = (float*)   (ws + off); off += (size_t)SEQ * BATCH * HID * 4;
  _Float16* w0h  = (_Float16*)(ws + off); off += (size_t)HID * DIN * 2;
  _Float16* w1h  = (_Float16*)(ws + off); off += (size_t)3 * HID * HID * 2;

  // f32 -> f16 staging (weights + input activations)
  cvt_f32_to_f16<<<1024, 256, 0, stream>>>(x,    xh,  SEQ * BATCH * DIN);
  cvt_f32_to_f16<<<512,  256, 0, stream>>>(Wih0, w0h, HID * DIN);
  cvt_f32_to_f16<<<1024, 256, 0, stream>>>(Wih,  w1h, 3 * HID * HID);

  const int M = SEQ * BATCH;
  dim3 gemm_grid(HID / BN, M / BM);          // (8, 512)
  dim3 gemm_blk(256);
  const int scan_blocks = (BATCH * HID) / 256;

  // layer 0: K = DIN
  wmma_gemm_f16<<<gemm_grid, gemm_blk, 0, stream>>>(xh, w0h, bih, pre, M, HID, DIN);
  indrnn_scan<<<scan_blocks, 256, 0, stream>>>(pre, whh, actA, nullptr, nullptr);

  // layer 1
  wmma_gemm_f16<<<gemm_grid, gemm_blk, 0, stream>>>(actA, w1h, bih + HID, pre, M, HID, HID);
  indrnn_scan<<<scan_blocks, 256, 0, stream>>>(pre, whh + HID, actB, nullptr, nullptr);

  // layer 2
  wmma_gemm_f16<<<gemm_grid, gemm_blk, 0, stream>>>(actB, w1h + (size_t)HID * HID,
                                                    bih + 2 * HID, pre, M, HID, HID);
  indrnn_scan<<<scan_blocks, 256, 0, stream>>>(pre, whh + 2 * HID, actA, nullptr, nullptr);

  // layer 3: write final out + h_last straight to d_out
  wmma_gemm_f16<<<gemm_grid, gemm_blk, 0, stream>>>(actA, w1h + 2 * (size_t)HID * HID,
                                                    bih + 3 * HID, pre, M, HID, HID);
  indrnn_scan<<<scan_blocks, 256, 0, stream>>>(pre, whh + 3 * HID, nullptr,
                                               out, out + (size_t)SEQ * BATCH * HID);
}
